// GlobalContextVitBlock_48163763258139
// MI455X (gfx1250) — matrix operand order repaired and checked by reading, not compiled, not run on hardware
//
#include <hip/hip_runtime.h>
#include <hip/hip_bf16.h>

// ---------------------------------------------------------------------------
// GCViT block for MI455X (gfx1250): bf16 WMMA everywhere, fused epilogues,
// async global->LDS staging (ASYNCcnt) for GEMM A tiles and attention K rows.
// ---------------------------------------------------------------------------

typedef __attribute__((ext_vector_type(16))) __bf16 v16bf;
typedef __attribute__((ext_vector_type(8)))  float  v8f;

union BFrag { v16bf v; unsigned int u[8]; unsigned short h[16]; };
union CFrag { v8f   v; float f[8]; };

__device__ __forceinline__ unsigned short f2bf(float f) {
  unsigned int u = __float_as_uint(f);
  unsigned int r = u + 0x7fffu + ((u >> 16) & 1u);   // round-to-nearest-even
  return (unsigned short)(r >> 16);
}

__device__ __forceinline__ v8f wmma_bf16(v16bf a, v16bf b, v8f c) {
  return __builtin_amdgcn_wmma_f32_16x16x32_bf16(false, a, false, b,
                                                 (short)0, c, false, false);
}

// async 16-byte global -> LDS copy (per active lane). INST_OFFSET is applied
// to both the LDS and global addresses per the ISA pseudocode.
__device__ __forceinline__ void async_b128(unsigned int lds_byte,
                                           unsigned long long gaddr) {
  asm volatile("global_load_async_to_lds_b128 %0, %1, off"
               :: "v"(lds_byte), "v"(gaddr) : "memory");
}

// ---------------------------------------------------------------------------
// fp32 -> bf16 bulk convert (weights)
// ---------------------------------------------------------------------------
__global__ void f2bf_kernel(const float* __restrict__ in,
                            unsigned short* __restrict__ out, int n) {
  int i = blockIdx.x * blockDim.x + threadIdx.x;
  if (i < n) out[i] = f2bf(in[i]);
}

// ---------------------------------------------------------------------------
// LayerNorm over C=384, one wave per row, bf16 output
// ---------------------------------------------------------------------------
__global__ __launch_bounds__(256)
void ln_kernel(const float* __restrict__ x, const float* __restrict__ g,
               const float* __restrict__ b, unsigned short* __restrict__ out,
               int T) {
  int row  = blockIdx.x * 8 + (threadIdx.x >> 5);
  int lane = threadIdx.x & 31;
  if (row >= T) return;
  const float* xr = x + (size_t)row * 384;
  float vals[12];
  float s = 0.f;
#pragma unroll
  for (int i = 0; i < 12; i++) { vals[i] = xr[lane + 32 * i]; s += vals[i]; }
#pragma unroll
  for (int off = 16; off > 0; off >>= 1) s += __shfl_xor(s, off, 32);
  float mean = s * (1.0f / 384.0f);
  float vs = 0.f;
#pragma unroll
  for (int i = 0; i < 12; i++) { float d = vals[i] - mean; vs += d * d; }
#pragma unroll
  for (int off = 16; off > 0; off >>= 1) vs += __shfl_xor(vs, off, 32);
  float inv = rsqrtf(vs * (1.0f / 384.0f) + 1e-5f);
#pragma unroll
  for (int i = 0; i < 12; i++) {
    int c = lane + 32 * i;
    out[(size_t)row * 384 + c] = f2bf((vals[i] - mean) * inv * g[c] + b[c]);
  }
}

// ---------------------------------------------------------------------------
// GEMM: out[T,N] = A[T,K](bf16) @ W[N,K](bf16)^T + bias, fused epilogue.
// Block = 128 thr (4 waves). Block tile: 16 rows x 128 cols; wave: 16x32.
// A tile staged via async global->LDS (no VGPR round-trip).
// EP: 0 = store bf16; 1 = xres + ls*val -> f32; 2 = GELU -> bf16;
//     3 = xres + ls*val -> f32 (with row offset, used for fc2/out)
// ---------------------------------------------------------------------------
template<int EP, int K>
__global__ __launch_bounds__(128)
void gemm_kernel(const unsigned short* __restrict__ A,
                 const unsigned short* __restrict__ W,
                 const float* __restrict__ bias,
                 const float* __restrict__ xres,
                 const float* __restrict__ ls,
                 void* __restrict__ outp, int out_ld, int row_off) {
  constexpr int KU = K / 2;                 // uints per row
  constexpr int CH = K / 8;                 // 16-byte chunks per row
  __shared__ __align__(16) unsigned int As[16 * KU];
  const int tid  = threadIdx.x;
  const int lane = tid & 31;
  const int wave = tid >> 5;
  const int rowbase = blockIdx.x * 16;
  const int colbase = blockIdx.y * 128 + wave * 32;

  // async-copy the 16xK bf16 A tile to LDS (16B per lane-op, uniform trips)
  for (int i = tid; i < 16 * CH; i += 128) {
    int r = i / CH, c = i - r * CH;
    async_b128((unsigned int)(unsigned long long)&As[r * KU + c * 4],
               (unsigned long long)(A + (size_t)(rowbase + r) * K + c * 8));
  }
  asm volatile("s_wait_asynccnt 0x0" ::: "memory");
  __syncthreads();

  const unsigned int* Wg = (const unsigned int*)W;
  const int m  = lane & 15;
  const int hi = lane >> 4;

  CFrag acc0, acc1;
#pragma unroll
  for (int r = 0; r < 8; r++) { acc0.f[r] = 0.f; acc1.f[r] = 0.f; }

  for (int kt = 0; kt < K / 32; kt++) {
    BFrag a;
#pragma unroll
    for (int j = 0; j < 8; j++) {   // ISA 16-bit A 16x32 layout
      int k = kt * 32 + ((j >> 2) << 4) + (hi << 3) + ((j & 3) << 1);
      a.u[j] = As[m * KU + (k >> 1)];
    }
    BFrag b0, b1;
#pragma unroll
    for (int j = 0; j < 8; j++) {   // ISA 16-bit B 32x16 layout
      int kk = kt * 32 + ((j >> 2) << 4) + (hi << 3) + ((j & 3) << 1);
      b0.u[j] = Wg[((size_t)(colbase + m)      * K + kk) >> 1];
      b1.u[j] = Wg[((size_t)(colbase + 16 + m) * K + kk) >> 1];
    }
    acc0.v = wmma_bf16(a.v, b0.v, acc0.v);
    acc1.v = wmma_bf16(a.v, b1.v, acc1.v);
  }

#pragma unroll
  for (int ci = 0; ci < 2; ci++) {
    CFrag& acc = ci ? acc1 : acc0;
    int col = colbase + ci * 16 + m;
    float bcol = bias[col];
#pragma unroll
    for (int r = 0; r < 8; r++) {
      int row   = rowbase + r + (hi << 3);
      float val = acc.f[r] + bcol;
      if (EP == 0) {
        ((unsigned short*)outp)[(size_t)row * out_ld + col] = f2bf(val);
      } else if (EP == 1 || EP == 3) {
        size_t gi = (size_t)(row + row_off) * out_ld + col;
        ((float*)outp)[gi] = xres[gi] + ls[col] * val;
      } else {  // EP == 2: exact GELU
        float gl = 0.5f * val * (1.0f + erff(val * 0.70710678118654752f));
        ((unsigned short*)outp)[(size_t)row * out_ld + col] = f2bf(gl);
      }
    }
  }
}

// ---------------------------------------------------------------------------
// Global-query window attention. One wave per (window, head).
// WIN=7, N=49 (padded to 64), HEADS=12, hd=32. 12288 tasks total.
// Per-wave LDS region (28928 B): [0..8K) q|k (later aliased by P bf16),
// [8K..12K) v^T, [12K..28K) S f32, [28K..28K+256) 1/rowsum.
// K rows are staged via async global->LDS; q (scale+convert) and v^T
// (transpose) go through the VALU path.
// ---------------------------------------------------------------------------
__global__ __launch_bounds__(64)
void attn_kernel(const float* __restrict__ qg,
                 const unsigned short* __restrict__ kv,
                 const float* __restrict__ bias_table,
                 unsigned short* __restrict__ o) {
  __shared__ __align__(16) char smem[2 * 28928];
  const int lane = threadIdx.x & 31;
  const int wib  = threadIdx.x >> 5;
  char* base = smem + wib * 28928;
  unsigned short* qS = (unsigned short*)(base);            // [64][32]
  unsigned short* kS = (unsigned short*)(base + 4096);     // [64][32]
  unsigned short* pS = (unsigned short*)(base);            // [64][64] alias q|k
  unsigned short* vT = (unsigned short*)(base + 8192);     // [32][64]
  float*          sS = (float*)(base + 12288);             // [64][64]
  float*        invS = (float*)(base + 28672);             // [64]

  const int task = blockIdx.x * 2 + wib;
  const int w = task / 12, h = task % 12;
  const int b  = w >> 6;            // 64 windows per image (8x8)
  const int wy = (w & 63) >> 3, wx = w & 7;
  const float scale = 0.17677669529663687f;  // 32^-0.5

  // ---- stage q (pre-scaled), k (async), v^T into LDS (bf16) ----
  for (int rr = 0; rr < 2; rr++) {
    int row = lane + rr * 32;
    if (row < 49) {
      int iy = row / 7, ix = row % 7;
      const float* qrow = qg + (((size_t)b * 7 + iy) * 7 + ix) * 384 + h * 32;
      size_t t = ((size_t)(b * 56 + wy * 7 + iy)) * 56 + (wx * 7 + ix);
      const unsigned short* kr = kv + t * 768 + h * 32;
      const unsigned short* vr = kv + t * 768 + 384 + h * 32;
      // k row: contiguous 64 B -> 4 async B128 copies straight into LDS
      unsigned int kl = (unsigned int)(unsigned long long)&kS[row * 32];
      unsigned long long kg = (unsigned long long)kr;
      asm volatile("global_load_async_to_lds_b128 %0, %1, off"
                   :: "v"(kl), "v"(kg) : "memory");
      asm volatile("global_load_async_to_lds_b128 %0, %1, off offset:16"
                   :: "v"(kl), "v"(kg) : "memory");
      asm volatile("global_load_async_to_lds_b128 %0, %1, off offset:32"
                   :: "v"(kl), "v"(kg) : "memory");
      asm volatile("global_load_async_to_lds_b128 %0, %1, off offset:48"
                   :: "v"(kl), "v"(kg) : "memory");
#pragma unroll
      for (int c = 0; c < 32; c++) {
        qS[row * 32 + c] = f2bf(qrow[c] * scale);
        vT[c * 64 + row] = vr[c];
      }
    } else {
#pragma unroll
      for (int c = 0; c < 32; c++) {
        qS[row * 32 + c] = 0; kS[row * 32 + c] = 0; vT[c * 64 + row] = 0;
      }
    }
  }
  asm volatile("s_wait_asynccnt 0x0" ::: "memory");
  asm volatile("s_wait_dscnt 0x0" ::: "memory");

  // ---- S = q @ k^T  (4x4 tiles of 16x16, K=32) ----
  const unsigned int* qU = (const unsigned int*)qS;
  const unsigned int* kU = (const unsigned int*)kS;
  const int m  = lane & 15;
  const int hi = lane >> 4;
  for (int mi = 0; mi < 4; mi++) {
    BFrag a;
#pragma unroll
    for (int j = 0; j < 8; j++) {
      int k = ((j >> 2) << 4) + (hi << 3) + ((j & 3) << 1);
      a.u[j] = qU[(mi * 16 + m) * 16 + (k >> 1)];
    }
    for (int ni = 0; ni < 4; ni++) {
      BFrag bf;
#pragma unroll
      for (int j = 0; j < 8; j++) {
        int kk = ((j >> 2) << 4) + (hi << 3) + ((j & 3) << 1);
        bf.u[j] = kU[(ni * 16 + m) * 16 + (kk >> 1)];
      }
      CFrag c;
#pragma unroll
      for (int r = 0; r < 8; r++) c.f[r] = 0.f;
      c.v = wmma_bf16(a.v, bf.v, c.v);
#pragma unroll
      for (int r = 0; r < 8; r++)
        sS[(mi * 16 + r + (hi << 3)) * 64 + ni * 16 + m] = c.f[r];
    }
  }
  asm volatile("s_wait_dscnt 0x0" ::: "memory");

  // ---- softmax rows (bias computed analytically); P unnormalized bf16 ----
  for (int rr = 0; rr < 2; rr++) {
    int row = lane + rr * 32;
    if (row < 49) {
      int qy = row / 7, qx = row % 7;
      float mx = -1e30f;
      for (int mm = 0; mm < 49; mm++) {
        int idx = (qy - mm / 7 + 6) * 13 + (qx - mm % 7 + 6);
        float v = sS[row * 64 + mm] + bias_table[idx * 12 + h];
        mx = fmaxf(mx, v);
      }
      float sum = 0.f;
      for (int mm = 0; mm < 49; mm++) {
        int idx = (qy - mm / 7 + 6) * 13 + (qx - mm % 7 + 6);
        float v = sS[row * 64 + mm] + bias_table[idx * 12 + h];
        float e = __expf(v - mx);
        sum += e;
        pS[row * 64 + mm] = f2bf(e);
      }
      for (int mm = 49; mm < 64; mm++) pS[row * 64 + mm] = 0;
      invS[row] = 1.0f / sum;
    } else {
      for (int mm = 0; mm < 64; mm++) pS[row * 64 + mm] = 0;
      invS[row] = 0.f;
    }
  }
  asm volatile("s_wait_dscnt 0x0" ::: "memory");

  // ---- O = P @ V  (4x2 tiles, K=64 in two 32-steps), scale by 1/rowsum ----
  const unsigned int* pU = (const unsigned int*)pS;
  const unsigned int* vU = (const unsigned int*)vT;
  for (int mi = 0; mi < 4; mi++) {
    for (int ni = 0; ni < 2; ni++) {
      CFrag c;
#pragma unroll
      for (int r = 0; r < 8; r++) c.f[r] = 0.f;
      for (int kt = 0; kt < 2; kt++) {
        BFrag a, bf;
#pragma unroll
        for (int j = 0; j < 8; j++) {
          int k = kt * 32 + ((j >> 2) << 4) + (hi << 3) + ((j & 3) << 1);
          a.u[j] = pU[(mi * 16 + m) * 32 + (k >> 1)];
        }
#pragma unroll
        for (int j = 0; j < 8; j++) {
          int kk = kt * 32 + ((j >> 2) << 4) + (hi << 3) + ((j & 3) << 1);
          bf.u[j] = vU[(ni * 16 + m) * 32 + (kk >> 1)];
        }
        c.v = wmma_bf16(a.v, bf.v, c.v);
      }
#pragma unroll
      for (int r = 0; r < 8; r++) {
        int row = mi * 16 + r + (hi << 3);
        if (row < 49) {
          int iy = row / 7, ix = row % 7;
          size_t t = ((size_t)(b * 56 + wy * 7 + iy)) * 56 + (wx * 7 + ix);
          float val = c.f[r] * invS[row];
          o[t * 384 + h * 32 + ni * 16 + m] = f2bf(val);
        }
      }
    }
  }
}

// ---------------------------------------------------------------------------
// Host orchestration
// ---------------------------------------------------------------------------
extern "C" void kernel_launch(void* const* d_in, const int* in_sizes, int n_in,
                              void* d_out, int out_size, void* d_ws, size_t ws_size,
                              hipStream_t stream) {
  const float* x       = (const float*)d_in[0];
  const float* qg      = (const float*)d_in[1];
  const float* ln1_g   = (const float*)d_in[2];
  const float* ln1_b   = (const float*)d_in[3];
  const float* qkv_w   = (const float*)d_in[4];
  const float* qkv_b   = (const float*)d_in[5];
  const float* proj_w  = (const float*)d_in[6];
  const float* proj_b  = (const float*)d_in[7];
  const float* bias_tb = (const float*)d_in[8];
  const float* ls1     = (const float*)d_in[9];
  const float* ln2_g   = (const float*)d_in[10];
  const float* ln2_b   = (const float*)d_in[11];
  const float* fc1_w   = (const float*)d_in[12];
  const float* fc1_b   = (const float*)d_in[13];
  const float* fc2_w   = (const float*)d_in[14];
  const float* fc2_b   = (const float*)d_in[15];
  const float* ls2     = (const float*)d_in[16];

  const int T = 50176;                       // 16*56*56
  char* ws = (char*)d_ws;
  size_t off = 0;
  unsigned short* kvb = (unsigned short*)(ws + off); off += 77070336;  // kv bf16 [T,768] (later h chunks)
  float*          x1  = (float*)         (ws + off); off += 77070336;  // residual fp32 [T,384]
  unsigned short* xnb = (unsigned short*)(ws + off); off += 38535168;  // ln out bf16 [T,384]
  unsigned short* ob  = (unsigned short*)(ws + off); off += 38535168;  // attn out bf16 [T,384]
  unsigned short* qkv_wb = (unsigned short*)(ws + off); off += 2u * 294912;
  unsigned short* proj_wb= (unsigned short*)(ws + off); off += 2u * 147456;
  unsigned short* fc1_wb = (unsigned short*)(ws + off); off += 2u * 589824;
  unsigned short* fc2_wb = (unsigned short*)(ws + off); off += 2u * 589824;
  unsigned short* hb = kvb;                  // reuse kv region for MLP hidden chunks

  // weights -> bf16 (L2-resident afterwards)
  f2bf_kernel<<<(294912 + 255) / 256, 256, 0, stream>>>(qkv_w,  qkv_wb,  294912);
  f2bf_kernel<<<(147456 + 255) / 256, 256, 0, stream>>>(proj_w, proj_wb, 147456);
  f2bf_kernel<<<(589824 + 255) / 256, 256, 0, stream>>>(fc1_w,  fc1_wb,  589824);
  f2bf_kernel<<<(589824 + 255) / 256, 256, 0, stream>>>(fc2_w,  fc2_wb,  589824);

  // LN1 -> bf16
  ln_kernel<<<T / 8, 256, 0, stream>>>(x, ln1_g, ln1_b, xnb, T);

  // kv = LN1(x) @ qkv_w^T + qkv_b   [T,768] bf16
  gemm_kernel<0, 384><<<dim3(T / 16, 6), 128, 0, stream>>>(
      xnb, qkv_wb, qkv_b, nullptr, nullptr, kvb, 768, 0);

  // window attention with global query -> o bf16 [T,384] (window_reverse fused)
  attn_kernel<<<6144, 64, 0, stream>>>(qg, kvb, bias_tb, ob);

  // x1 = x + ls1 * (o @ proj_w^T + proj_b)
  gemm_kernel<1, 384><<<dim3(T / 16, 3), 128, 0, stream>>>(
      ob, proj_wb, proj_b, x, ls1, x1, 384, 0);

  // LN2 -> bf16 (reuse xnb)
  ln_kernel<<<T / 8, 256, 0, stream>>>(x1, ln2_g, ln2_b, xnb, T);

  // MLP in 4 token chunks (hidden reuses kv region)
  for (int cidx = 0; cidx < 4; cidx++) {
    int t0 = cidx * 12544;
    gemm_kernel<2, 384><<<dim3(12544 / 16, 12), 128, 0, stream>>>(
        xnb + (size_t)t0 * 384, fc1_wb, fc1_b, nullptr, nullptr, hb, 1536, 0);
    gemm_kernel<3, 1536><<<dim3(12544 / 16, 3), 128, 0, stream>>>(
        hb, fc2_wb, fc2_b, x1, ls2, d_out, 384, t0);
  }
  (void)in_sizes; (void)n_in; (void)out_size; (void)ws_size;
}
